// ONNXCompatibleMultiHeadAttention_43490838839603
// MI455X (gfx1250) — compile-verified, matrix operand order checked by reading, not run on hardware
//
#include <hip/hip_runtime.h>

#define B_  2
#define T_  2048
#define S_  2048
#define E_  1024
#define H_  16
#define DH_ 64

typedef __attribute__((ext_vector_type(16))) _Float16 v16h;
typedef __attribute__((ext_vector_type(8)))  _Float16 v8h;
typedef __attribute__((ext_vector_type(8)))  float    v8f;

static __device__ __forceinline__ v16h joinv(v8h a, v8h b) {
    return __builtin_shufflevector(a, b, 0,1,2,3,4,5,6,7,8,9,10,11,12,13,14,15);
}
static __device__ __forceinline__ v8f wmma_f16(v16h a, v16h b, v8f c) {
    // v_wmma_f32_16x16x32_f16: D = A(16x32 f16) * B(32x16 f16) + C(16x16 f32)
    return __builtin_amdgcn_wmma_f32_16x16x32_f16(false, a, false, b, (short)0, c,
                                                  false, false);
}

// ---------------------------------------------------------------------------
// C = X[4096,1024] @ W[1024,1024]^T + bias, f16 WMMA with f32 accumulate.
// MODE 0: half out, head-split [B,H,T,DH]             (Q, K)
// MODE 1: half out, head-split transposed [B,H,DH,S]  (V)
// MODE 2: float out, row-major [M,N]                  (final projection)
// ---------------------------------------------------------------------------
template<int MODE>
__global__ __launch_bounds__(256) void gemm_xWt(const float* __restrict__ X,
                                                const float* __restrict__ W,
                                                const float* __restrict__ bias,
                                                void* __restrict__ outp) {
    __shared__ _Float16 lA[128][40];  // 128 x 32 tile of X, padded rows
    __shared__ _Float16 lB[64][40];   // 64 x 32 tile of W rows (B = W^T cols)

    const int tid  = threadIdx.x;
    const int lane = tid & 31;
    const int wid  = tid >> 5;                 // 8 waves
    const int m0   = blockIdx.y * 128;
    const int n0   = blockIdx.x * 64;
    const int ln   = lane & 15;
    const int lh   = lane >> 4;

    v8f acc[4] = {};

    for (int k0 = 0; k0 < E_; k0 += 32) {
        { // cooperative A-tile load: 128 rows x 32 cols, f32 -> f16
            const int r = tid >> 1, c = (tid & 1) * 16;
            const float* src = X + (size_t)(m0 + r) * E_ + k0 + c;
            #pragma unroll
            for (int i = 0; i < 16; ++i) lA[r][c + i] = (_Float16)src[i];
            if (k0 + 32 < E_) __builtin_prefetch(src + 32, 0, 1); // global_prefetch_b8
        }
        { // cooperative B-tile load: 64 W-rows x 32 cols
            const int r = tid >> 2, c = (tid & 3) * 8;
            const float* src = W + (size_t)(n0 + r) * E_ + k0 + c;
            #pragma unroll
            for (int i = 0; i < 8; ++i) lB[r][c + i] = (_Float16)src[i];
        }
        __syncthreads();

        const int ar = wid * 16 + ln;
        const int kk = lh * 16;
        v16h a = joinv(*(const v8h*)&lA[ar][kk], *(const v8h*)&lA[ar][kk + 8]);
        #pragma unroll
        for (int nt = 0; nt < 4; ++nt) {
            const int br = nt * 16 + ln;
            v16h b = joinv(*(const v8h*)&lB[br][kk], *(const v8h*)&lB[br][kk + 8]);
            acc[nt] = wmma_f16(a, b, acc[nt]);
        }
        __syncthreads();
    }

    // Epilogue: C tile element (m,n): m = r + 8*(lane>>4), n = lane&15.
    #pragma unroll
    for (int nt = 0; nt < 4; ++nt) {
        #pragma unroll
        for (int r = 0; r < 8; ++r) {
            const int gm = m0 + wid * 16 + r + 8 * lh;
            const int gn = n0 + nt * 16 + ln;
            const float vv = acc[nt][r] + bias[gn];
            if (MODE == 0) {
                const int b = gm >> 11, t = gm & (T_ - 1);
                const int h = gn >> 6,  d = gn & (DH_ - 1);
                ((_Float16*)outp)[((size_t)(b * H_ + h) * T_ + t) * DH_ + d] =
                    (_Float16)vv;
            } else if (MODE == 1) {
                const int b = gm >> 11, s = gm & (S_ - 1);
                const int h = gn >> 6,  d = gn & (DH_ - 1);
                ((_Float16*)outp)[((size_t)(b * H_ + h) * DH_ + d) * S_ + s] =
                    (_Float16)vv;
            } else {
                ((float*)outp)[(size_t)gm * E_ + gn] = vv;
            }
        }
    }
}

// ---------------------------------------------------------------------------
// One 512-thread block = 16 waves = 16 heads, per (b, 16-row t-tile).
// Flash-style two-sweep softmax per wave; per-lane online (m,l) with a single
// end-of-sweep butterfly merge. Head-averaged attention accumulates in a
// 16x2048 f32 LDS tile via ds_add_f32, then streams to HBM with
// global_store_async_from_lds_b128 (no VGPR round trip, no global atomics).
// Dynamic LDS: 16*2048*4 (avg) + 16*16*40*2 (P staging) = 151552 bytes.
// ---------------------------------------------------------------------------
__global__ __launch_bounds__(512) void attn_fwd(const _Float16* __restrict__ Qh,
                                                const _Float16* __restrict__ Kh,
                                                const _Float16* __restrict__ Vt,
                                                float* __restrict__ ctx,
                                                float* __restrict__ attn_avg) {
    extern __shared__ char smem[];
    float*    lAvg  = (float*)smem;                                // [16][2048]
    _Float16* lPall = (_Float16*)(smem + (size_t)16 * S_ * sizeof(float));

    const int tid  = threadIdx.x;
    const int lane = tid & 31;
    const int h    = tid >> 5;                 // wave id == head index
    const int ln   = lane & 15, lh = lane >> 4;
    const int t0   = blockIdx.x * 16;
    const int b    = blockIdx.y;
    const size_t bh = (size_t)(b * H_ + h);
    const float scale = 0.125f;                // 1/sqrt(DH)
    _Float16* lP = lPall + h * (16 * 40);      // per-wave [16][40]

    { // zero the head-average accumulator (block-wide)
        float4* z = (float4*)lAvg;
        #pragma unroll
        for (int i = 0; i < 16; ++i)
            z[tid + i * 512] = make_float4(0.f, 0.f, 0.f, 0.f);
    }
    __syncthreads();

    // Q A-fragments (16 x 64 = two 16x32 k-steps), rows contiguous per lane.
    const _Float16* qp = Qh + (bh * T_ + t0 + ln) * DH_ + lh * 16;
    const v16h aq0 = *(const v16h*)qp;
    const v16h aq1 = *(const v16h*)(qp + 32);

    // Per-lane local row state over this lane's score columns.
    // Lanes 0-15 own rows 0-7, lanes 16-31 own rows 8-15 (C-matrix layout).
    float mrow[8], lrow[8];
    #pragma unroll
    for (int r = 0; r < 8; ++r) { mrow[r] = -1e30f; lrow[r] = 0.f; }

    // ---- Sweep 1: per-lane online max/expsum, no cross-lane traffic ----
    for (int s0 = 0; s0 < S_; s0 += 16) {
        const _Float16* kp = Kh + (bh * S_ + s0 + ln) * DH_ + lh * 16;
        v8f c = {};
        c = wmma_f16(aq0, *(const v16h*)kp, c);
        c = wmma_f16(aq1, *(const v16h*)(kp + 32), c);
        #pragma unroll
        for (int r = 0; r < 8; ++r) {
            const float v  = c[r] * scale;
            const float nm = fmaxf(mrow[r], v);
            lrow[r] = lrow[r] * __expf(mrow[r] - nm) + __expf(v - nm);
            mrow[r] = nm;
        }
    }
    // Single butterfly merge of (m,l) across the 16 lanes of each half-wave.
    #pragma unroll
    for (int r = 0; r < 8; ++r) {
        #pragma unroll
        for (int o = 1; o < 16; o <<= 1) {
            const float mo = __shfl_xor(mrow[r], o, 32);
            const float lo = __shfl_xor(lrow[r], o, 32);
            const float nm = fmaxf(mrow[r], mo);
            lrow[r] = lrow[r] * __expf(mrow[r] - nm) + lo * __expf(mo - nm);
            mrow[r] = nm;
        }
    }
    float linv[8];
    #pragma unroll
    for (int r = 0; r < 8; ++r) linv[r] = 1.0f / lrow[r];

    // ---- Sweep 2: P = softmax row, O += P @ V, lAvg += P/H (LDS atomics) ----
    v8f o[4] = {};
    for (int s0 = 0; s0 < S_; s0 += 32) {
        #pragma unroll
        for (int st = 0; st < 2; ++st) {
            const int s = s0 + st * 16;
            const _Float16* kp = Kh + (bh * S_ + s + ln) * DH_ + lh * 16;
            v8f c = {};
            c = wmma_f16(aq0, *(const v16h*)kp, c);
            c = wmma_f16(aq1, *(const v16h*)(kp + 32), c);
            #pragma unroll
            for (int r = 0; r < 8; ++r) {
                const int m = r + 8 * lh;
                const float p = __expf(c[r] * scale - mrow[r]) * linv[r];
                lP[m * 40 + st * 16 + ln] = (_Float16)p;
                atomicAdd(&lAvg[m * S_ + s + ln], p * (1.0f / H_)); // ds_add_f32
            }
        }
        // Wave-local DS ordering: staging writes -> fragment reads (same wave).
        asm volatile("s_wait_dscnt 0x0" ::: "memory");
        // P as A-fragment (16 x 32 over s); V^T rows give contiguous B-frags.
        v16h ap = joinv(*(const v8h*)&lP[ln * 40 + lh * 16],
                        *(const v8h*)&lP[ln * 40 + lh * 16 + 8]);
        #pragma unroll
        for (int nt = 0; nt < 4; ++nt) {
            const _Float16* vp = Vt + (bh * DH_ + nt * 16 + ln) * S_ + s0 + lh * 16;
            o[nt] = wmma_f16(ap, *(const v16h*)vp, o[nt]);
        }
        asm volatile("s_wait_dscnt 0x0" ::: "memory");
    }

    // ctx[b, t, h*64 + d] = O
    #pragma unroll
    for (int nt = 0; nt < 4; ++nt) {
        #pragma unroll
        for (int r = 0; r < 8; ++r) {
            const int t = t0 + r + 8 * lh;
            const int e = h * DH_ + nt * 16 + ln;
            ctx[((size_t)b * T_ + t) * E_ + e] = o[nt][r];
        }
    }

    // ---- Stream the finished 16x2048 average tile LDS -> HBM (async) ----
    __syncthreads();
    {
        const int row = tid >> 5;  // 16 rows, one wave per row
        const size_t grow = ((size_t)b * T_ + t0 + row) * S_;
        #pragma unroll
        for (int cc = 0; cc < 16; ++cc) {
            const int col = cc * 128 + lane * 4;  // float4 per lane, coalesced
            const unsigned lds_addr =
                (unsigned)(uintptr_t)&lAvg[row * S_ + col];
            const unsigned long long gaddr =
                (unsigned long long)(uintptr_t)(attn_avg + grow + col);
            asm volatile("global_store_async_from_lds_b128 %0, %1, off"
                         :: "v"(gaddr), "v"(lds_addr) : "memory");
        }
        asm volatile("s_wait_asynccnt 0x0" ::: "memory");
    }
}

extern "C" void kernel_launch(void* const* d_in, const int* in_sizes, int n_in,
                              void* d_out, int out_size, void* d_ws, size_t ws_size,
                              hipStream_t stream) {
    const float* q  = (const float*)d_in[0];
    const float* k  = (const float*)d_in[1];
    const float* v  = (const float*)d_in[2];
    const float* Wq = (const float*)d_in[3];
    const float* bq = (const float*)d_in[4];
    const float* Wk = (const float*)d_in[5];
    const float* bk = (const float*)d_in[6];
    const float* Wv = (const float*)d_in[7];
    const float* bv = (const float*)d_in[8];
    const float* Wo = (const float*)d_in[9];
    const float* bo = (const float*)d_in[10];

    // Workspace: Qh, Kh (f16, [B,H,T,DH]), Vt (f16, [B,H,DH,S]), ctx (f32).
    const size_t nQKV = (size_t)B_ * H_ * T_ * DH_;  // 4 Mi elements each
    _Float16* Qh = (_Float16*)d_ws;
    _Float16* Kh = Qh + nQKV;
    _Float16* Vt = Kh + nQKV;
    float*    ctx = (float*)(Vt + nQKV);

    float* outp     = (float*)d_out;
    float* attn_avg = outp + (size_t)B_ * T_ * E_;

    dim3 gg(E_ / 64, (B_ * T_) / 128);
    gemm_xWt<0><<<gg, 256, 0, stream>>>(q, Wq, bq, Qh);
    gemm_xWt<0><<<gg, 256, 0, stream>>>(k, Wk, bk, Kh);
    gemm_xWt<1><<<gg, 256, 0, stream>>>(v, Wv, bv, Vt);

    const size_t smem = (size_t)16 * S_ * sizeof(float)      // avg accumulator
                      + (size_t)H_ * 16 * 40 * sizeof(_Float16);  // P staging
    attn_fwd<<<dim3(T_ / 16, B_), 512, smem, stream>>>(Qh, Kh, Vt, ctx, attn_avg);

    gemm_xWt<2><<<gg, 256, 0, stream>>>(ctx, Wo, bo, outp);
}
